// EGNN_Sparse_Network_11330123727317
// MI455X (gfx1250) — matrix operand order repaired and checked by reading, not compiled, not run on hardware
//
#include <hip/hip_runtime.h>
#include <hip/hip_bf16.h>
#include <cstdint>

// ---------------------------------------------------------------------------
// Types for CDNA5 WMMA (wave32): D(16x16 f32) = A(16x32 bf16) * B(32x16 bf16) + C
// ---------------------------------------------------------------------------
typedef __bf16 bf16;
typedef __attribute__((ext_vector_type(16))) __bf16 v16bf;
typedef __attribute__((ext_vector_type(8)))  __bf16 v8bf;
typedef __attribute__((ext_vector_type(8)))  float  v8f;

#define WMMA_BF16(a, b, c) \
  __builtin_amdgcn_wmma_f32_16x16x32_bf16(false, (a), false, (b), (short)0, (c), false, false)

__device__ __forceinline__ float silu(float x) {
  return x / (1.0f + __expf(-x));
}

// A-fragment (16x32, bf16) from a row-major LDS tile.
// lane 0..15  (M=lane)    : K = {0..7} and {16..23}
// lane 16..31 (M=lane-16) : K = {8..15} and {24..31}
// rowbase points at lds_row_base + ktile*32 ; grp = lane>>4
__device__ __forceinline__ v16bf load_a_frag(const bf16* rowbase, int grp) {
  const v8bf lo = *(const v8bf*)(rowbase + grp * 8);
  const v8bf hi = *(const v8bf*)(rowbase + grp * 8 + 16);
  v16bf a;
#pragma unroll
  for (int i = 0; i < 8; ++i) { a[i] = lo[i]; a[i + 8] = hi[i]; }
  return a;
}

// B-fragment (32x16, bf16) from pre-transposed weights Wt[n][k] (K-contiguous).
// lane = column n (mod 16); lanes 0-15 hold K={0..7,16..23}, lanes 16-31 hold K={8..15,24..31}.
__device__ __forceinline__ v16bf load_b_frag(const bf16* colbase, int grp) {
  const v8bf lo = *(const v8bf*)(colbase + grp * 8);
  const v8bf hi = *(const v8bf*)(colbase + grp * 8 + 16);
  v16bf b;
#pragma unroll
  for (int i = 0; i < 8; ++i) { b[i] = lo[i]; b[i + 8] = hi[i]; }
  return b;
}

// ---------------------------------------------------------------------------
// Problem constants (reference: EGNN, 3 layers)
// ---------------------------------------------------------------------------
#define XSTR  131        // POS(3) + FEATS(128)
#define FE    128        // FEATS
#define EINK  257        // 1 + 2*FEATS
#define KP1   288        // EINK padded to mult of 32
#define N1    514        // 2*E_IN
#define NT1   33         // ceil(514/16)
#define KP2   544        // N1 padded to mult of 32
#define MD    64         // M_DIM
#define KN1   192        // FEATS + M_DIM
#define KN2   256        // 2*FEATS

// ---------------------------------------------------------------------------
// Prep: fp32 [K][N] weight  ->  bf16 [Npad][Kpad] (transposed, zero padded)
// ---------------------------------------------------------------------------
__global__ void egnn_prep_wT(const float* __restrict__ src, bf16* __restrict__ dst,
                             int K, int N, int Kpad, int Npad) {
  int idx = blockIdx.x * blockDim.x + threadIdx.x;
  int total = Npad * Kpad;
  if (idx >= total) return;
  int n = idx / Kpad, k = idx - n * Kpad;
  float v = (n < N && k < K) ? src[(long)k * N + n] : 0.0f;
  dst[idx] = (bf16)v;
}

__global__ void egnn_zero_f32(float* __restrict__ p, long n) {
  long idx = (long)blockIdx.x * blockDim.x + threadIdx.x;
  if (idx < n) p[idx] = 0.0f;
}

// ---------------------------------------------------------------------------
// Edge kernel: one wave per 32-edge group (2 M-tiles of 16) so every streamed
// B fragment feeds two WMMAs. GEMM2 is fused into the GEMM1 n-loop through a
// rolling 16x32 h-chunk in LDS (never materialize full h).
// Two waves per block; all tiles wave-private -> no block barriers.
// ---------------------------------------------------------------------------
__global__ __launch_bounds__(64) void egnn_edge_kernel(
    const float* __restrict__ xin, const int* __restrict__ ei, int nEdges,
    const bf16* __restrict__ w1t, const float* __restrict__ b1,
    const bf16* __restrict__ w2t, const float* __restrict__ b2,
    const bf16* __restrict__ cw1t, const float* __restrict__ cb1,
    const float* __restrict__ cw2, const float* __restrict__ cb2,
    float* __restrict__ m_i, float* __restrict__ mhat) {
  __shared__ __align__(16) bf16  sh_e[2][2][16 * KP1];   // e_in tiles
  __shared__ __align__(16) bf16  sh_hc[2][2][16 * 32];   // rolling h K-chunk
  __shared__ __align__(16) bf16  sh_m[2][2][16 * MD];    // m_ij tiles
  __shared__ __align__(16) float sh_rel[2][2][16 * 3];
  __shared__ __align__(16) float sh_rd[2][2][16];
  __shared__ __align__(16) float sh_cw[2][2][16];
  __shared__ __align__(16) int   sh_src[2][2][16];
  __shared__ __align__(16) int   sh_dst[2][2][16];

  const int wave  = threadIdx.x >> 5;
  const int lane  = threadIdx.x & 31;
  const int row16 = lane & 15;
  const int grp   = lane >> 4;
  const int e0    = (blockIdx.x * 2 + wave) * 32;   // 32 edges per wave
  if (e0 >= nEdges) return;

  // ---- per-edge geometry: all 32 lanes, lane -> (tile=grp, row=row16) ----
  {
    const int e = e0 + lane;
    const int s = ei[e];
    const int d = ei[nEdges + e];
    float rx = xin[(long)s * XSTR + 0] - xin[(long)d * XSTR + 0];
    float ry = xin[(long)s * XSTR + 1] - xin[(long)d * XSTR + 1];
    float rz = xin[(long)s * XSTR + 2] - xin[(long)d * XSTR + 2];
    sh_rel[wave][grp][row16 * 3 + 0] = rx;
    sh_rel[wave][grp][row16 * 3 + 1] = ry;
    sh_rel[wave][grp][row16 * 3 + 2] = rz;
    sh_rd[wave][grp][row16]  = rx * rx + ry * ry + rz * rz;
    sh_src[wave][grp][row16] = s;
    sh_dst[wave][grp][row16] = d;
  }

  // ---- stage e_in tiles (2 x 16 x 288 bf16): [feats[dst]|feats[src]|dist|0]
  for (int rr = 0; rr < 32; ++rr) {
    const int t = rr >> 4, r = rr & 15;
    const long db = (long)sh_dst[wave][t][r] * XSTR + 3;
    const long sb = (long)sh_src[wave][t][r] * XSTR + 3;
    bf16* lds_e = sh_e[wave][t];
#pragma unroll
    for (int i = 0; i < KP1 / 32; ++i) {
      int c = i * 32 + lane;
      float v;
      if (c < FE)            v = xin[db + c];
      else if (c < 2 * FE)   v = xin[sb + (c - FE)];
      else if (c == 2 * FE)  v = sh_rd[wave][t][r];
      else                   v = 0.0f;
      lds_e[r * KP1 + c] = (bf16)v;
    }
  }

  // ---- fused GEMM1 + GEMM2 ----
  // GEMM1: h = silu(e_in @ ew1 + eb1), produced 32 columns (one K-chunk) at a
  // time into sh_hc, immediately consumed by GEMM2: m_ij = h @ ew2.
  const v8f vzero = {0.f, 0.f, 0.f, 0.f, 0.f, 0.f, 0.f, 0.f};
  v8f macc[2][4];
#pragma unroll
  for (int t = 0; t < 2; ++t)
#pragma unroll
    for (int j = 0; j < 4; ++j) macc[t][j] = vzero;

  for (int p = 0; p < KP2 / 32; ++p) {            // 17 K-chunks of GEMM2
#pragma unroll
    for (int half = 0; half < 2; ++half) {        // two GEMM1 n-tiles per chunk
      const int nt = 2 * p + half;
      if (nt < NT1) {
        const int n = nt * 16 + row16;
        const bf16* bcol = w1t + (long)n * KP1;
        __builtin_prefetch(w1t + (long)(n + 16) * KP1, 0, 1);
        v8f a0c = vzero, a1c = vzero;
#pragma unroll
        for (int kt = 0; kt < KP1 / 32; ++kt) {
          v16bf b  = load_b_frag(bcol + kt * 32, grp);
          v16bf a0 = load_a_frag(sh_e[wave][0] + row16 * KP1 + kt * 32, grp);
          a0c = WMMA_BF16(a0, b, a0c);
          v16bf a1 = load_a_frag(sh_e[wave][1] + row16 * KP1 + kt * 32, grp);
          a1c = WMMA_BF16(a1, b, a1c);
        }
        const float bias = (n < N1) ? b1[n] : 0.0f;
#pragma unroll
        for (int r = 0; r < 8; ++r) {
          const int mrow = grp * 8 + r;
          sh_hc[wave][0][mrow * 32 + half * 16 + row16] = (bf16)silu(a0c[r] + bias);
          sh_hc[wave][1][mrow * 32 + half * 16 + row16] = (bf16)silu(a1c[r] + bias);
        }
      } else {
        // nt == 33: zero-fill the second half of the final chunk
#pragma unroll
        for (int i = 0; i < 8; ++i) {
          int idx = i * 32 + lane;                 // 16x16 block
          int r = idx >> 4, c = idx & 15;
          sh_hc[wave][0][r * 32 + half * 16 + c] = (bf16)0.0f;
          sh_hc[wave][1][r * 32 + half * 16 + c] = (bf16)0.0f;
        }
      }
    }
    // consume this K-chunk into the four GEMM2 accumulators (per tile)
    v16bf a0 = load_a_frag(sh_hc[wave][0] + row16 * 32, grp);
    v16bf a1 = load_a_frag(sh_hc[wave][1] + row16 * 32, grp);
#pragma unroll
    for (int nt2 = 0; nt2 < 4; ++nt2) {
      const int n2 = nt2 * 16 + row16;
      v16bf b = load_b_frag(w2t + (long)n2 * KP2 + p * 32, grp);
      macc[0][nt2] = WMMA_BF16(a0, b, macc[0][nt2]);
      macc[1][nt2] = WMMA_BF16(a1, b, macc[1][nt2]);
    }
  }

  // ---- m_ij epilogue: bias + silu, stage for coor gate, scatter into m_i ----
#pragma unroll
  for (int t = 0; t < 2; ++t) {
#pragma unroll
    for (int nt2 = 0; nt2 < 4; ++nt2) {
      const int n2 = nt2 * 16 + row16;
      const float bias = b2[n2];
#pragma unroll
      for (int r = 0; r < 8; ++r) {
        const int mrow = grp * 8 + r;
        const float mv = silu(macc[t][nt2][r] + bias);
        sh_m[wave][t][mrow * MD + n2] = (bf16)mv;
        const int d = sh_dst[wave][t][mrow];
        atomicAdd(&m_i[(long)d * MD + n2], mv);
      }
    }
  }

  // ---- coordinate gate: coor_w = silu(m_ij @ cw1 + cb1) @ cw2 + cb2 ----
  v8f cacc0 = vzero, cacc1 = vzero;
  for (int nt = 0; nt < 16; ++nt) {
    const int n = nt * 16 + row16;
    const bf16* bcol = cw1t + (long)n * MD;
    v8f q0 = vzero, q1 = vzero;
#pragma unroll
    for (int kt = 0; kt < MD / 32; ++kt) {
      v16bf b  = load_b_frag(bcol + kt * 32, grp);
      v16bf a0 = load_a_frag(sh_m[wave][0] + row16 * MD + kt * 32, grp);
      q0 = WMMA_BF16(a0, b, q0);
      v16bf a1 = load_a_frag(sh_m[wave][1] + row16 * MD + kt * 32, grp);
      q1 = WMMA_BF16(a1, b, q1);
    }
    const float bias = cb1[n];
    const float w2   = cw2[n];
#pragma unroll
    for (int r = 0; r < 8; ++r) {
      cacc0[r] += silu(q0[r] + bias) * w2;
      cacc1[r] += silu(q1[r] + bias) * w2;
    }
  }
  // row-sum across the 16 lanes holding the N columns
#pragma unroll
  for (int r = 0; r < 8; ++r) {
    float v0 = cacc0[r], v1 = cacc1[r];
    v0 += __shfl_xor(v0, 1, 16); v1 += __shfl_xor(v1, 1, 16);
    v0 += __shfl_xor(v0, 2, 16); v1 += __shfl_xor(v1, 2, 16);
    v0 += __shfl_xor(v0, 4, 16); v1 += __shfl_xor(v1, 4, 16);
    v0 += __shfl_xor(v0, 8, 16); v1 += __shfl_xor(v1, 8, 16);
    cacc0[r] = v0; cacc1[r] = v1;
  }
  if (row16 == 0) {
    const float c2 = cb2[0];
#pragma unroll
    for (int r = 0; r < 8; ++r) {
      sh_cw[wave][0][grp * 8 + r] = cacc0[r] + c2;
      sh_cw[wave][1][grp * 8 + r] = cacc1[r] + c2;
    }
  }
  // ---- scatter coor_w * rel_coors into mhat[dst]; lane -> edge (grp,row16)
  {
    const float w = sh_cw[wave][grp][row16];
    const int   d = sh_dst[wave][grp][row16];
#pragma unroll
    for (int k = 0; k < 3; ++k)
      atomicAdd(&mhat[(long)d * 3 + k], w * sh_rel[wave][grp][row16 * 3 + k]);
  }
}

// ---------------------------------------------------------------------------
// Node kernel: LayerNorm + node MLP (two WMMA GEMMs) + residuals + coors
// ---------------------------------------------------------------------------
__global__ __launch_bounds__(64) void egnn_node_kernel(
    const float* __restrict__ xin, const float* __restrict__ m_i,
    const float* __restrict__ mhat,
    const bf16* __restrict__ nw1t, const float* __restrict__ nb1,
    const bf16* __restrict__ nw2t, const float* __restrict__ nb2,
    const float* __restrict__ ng, const float* __restrict__ nb,
    float* __restrict__ xout, int nNodes) {
  __shared__ __align__(16) bf16 sh_n[2][16 * KN1];
  __shared__ __align__(16) bf16 sh_q[2][16 * KN2];

  const int wave  = threadIdx.x >> 5;
  const int lane  = threadIdx.x & 31;
  const int row16 = lane & 15;
  const int grp   = lane >> 4;
  const int node0 = (blockIdx.x * 2 + wave) * 16;
  if (node0 >= nNodes) return;

  bf16* lds_n = sh_n[wave];
  bf16* lds_q = sh_q[wave];

  // ---- LayerNorm(feats) and staging of n_in = [feats_n | m_i] (16 x 192) ----
  for (int r = 0; r < 16; ++r) {
    const long base = (long)(node0 + r) * XSTR + 3;
    float v[4], s = 0.f, sq = 0.f;
#pragma unroll
    for (int j = 0; j < 4; ++j) {
      v[j] = xin[base + j * 32 + lane];
      s += v[j];
      sq += v[j] * v[j];
    }
#pragma unroll
    for (int m = 1; m < 32; m <<= 1) {
      s  += __shfl_xor(s, m, 32);
      sq += __shfl_xor(sq, m, 32);
    }
    const float mu   = s * (1.0f / 128.0f);
    const float var  = sq * (1.0f / 128.0f) - mu * mu;
    const float rstd = rsqrtf(var + 1e-5f);
#pragma unroll
    for (int j = 0; j < 4; ++j) {
      int c = j * 32 + lane;
      lds_n[r * KN1 + c] = (bf16)((v[j] - mu) * rstd * ng[c] + nb[c]);
    }
#pragma unroll
    for (int j = 0; j < 2; ++j) {
      int c = j * 32 + lane;
      lds_n[r * KN1 + FE + c] = (bf16)m_i[(long)(node0 + r) * MD + c];
    }
  }

  // ---- GEMM1: q = silu(n_in @ nw1 + nb1)   (16x192) x (192x256) ----
  for (int nt = 0; nt < 16; ++nt) {
    v8f acc = {0.f, 0.f, 0.f, 0.f, 0.f, 0.f, 0.f, 0.f};
    const int n = nt * 16 + row16;
    const bf16* bcol = nw1t + (long)n * KN1;
    const bf16* arow = lds_n + row16 * KN1;
#pragma unroll
    for (int kt = 0; kt < KN1 / 32; ++kt) {
      v16bf a = load_a_frag(arow + kt * 32, grp);
      v16bf b = load_b_frag(bcol + kt * 32, grp);
      acc = WMMA_BF16(a, b, acc);
    }
    const float bias = nb1[n];
#pragma unroll
    for (int r = 0; r < 8; ++r)
      lds_q[(grp * 8 + r) * KN2 + n] = (bf16)silu(acc[r] + bias);
  }

  // ---- GEMM2: feats_out = feats + q @ nw2 + nb2   (16x256) x (256x128) ----
  for (int nt = 0; nt < 8; ++nt) {
    v8f acc = {0.f, 0.f, 0.f, 0.f, 0.f, 0.f, 0.f, 0.f};
    const int n = nt * 16 + row16;
    const bf16* bcol = nw2t + (long)n * KN2;
    const bf16* arow = lds_q + row16 * KN2;
#pragma unroll
    for (int kt = 0; kt < KN2 / 32; ++kt) {
      v16bf a = load_a_frag(arow + kt * 32, grp);
      v16bf b = load_b_frag(bcol + kt * 32, grp);
      acc = WMMA_BF16(a, b, acc);
    }
    const float bias = nb2[n];
#pragma unroll
    for (int r = 0; r < 8; ++r) {
      const int  m = grp * 8 + r;
      const long o = (long)(node0 + m) * XSTR + 3 + n;
      xout[o] = xin[o] + acc[r] + bias;
    }
  }

  // ---- coors_out = coors + mhat ----
  if (lane < 16) {
    const int node = node0 + lane;
#pragma unroll
    for (int k = 0; k < 3; ++k)
      xout[(long)node * XSTR + k] = xin[(long)node * XSTR + k] + mhat[(long)node * 3 + k];
  }
}

// ---------------------------------------------------------------------------
// Host launcher
// ---------------------------------------------------------------------------
extern "C" void kernel_launch(void* const* d_in, const int* in_sizes, int n_in,
                              void* d_out, int out_size, void* d_ws, size_t ws_size,
                              hipStream_t stream) {
  const float* x   = (const float*)d_in[0];
  const int*   ei  = (const int*)d_in[1];
  const float* ew1 = (const float*)d_in[2];
  const float* eb1 = (const float*)d_in[3];
  const float* ew2 = (const float*)d_in[4];
  const float* eb2 = (const float*)d_in[5];
  const float* ng  = (const float*)d_in[6];
  const float* nb  = (const float*)d_in[7];
  const float* nw1 = (const float*)d_in[8];
  const float* nb1 = (const float*)d_in[9];
  const float* nw2 = (const float*)d_in[10];
  const float* nb2 = (const float*)d_in[11];
  const float* cw1 = (const float*)d_in[12];
  const float* cb1 = (const float*)d_in[13];
  const float* cw2 = (const float*)d_in[14];
  const float* cb2 = (const float*)d_in[15];

  const int nNodes = in_sizes[0] / XSTR;   // 20000
  const int nEdges = in_sizes[1] / 2;      // 320000
  const int L = 3;

  // ---- workspace carve-up (256B aligned) ----
  char* base = (char*)d_ws;
  size_t off = 0;
  auto carve = [&](size_t bytes) -> char* {
    char* p = base + off;
    off = (off + bytes + 255) & ~(size_t)255;
    return p;
  };
  float* bufA  = (float*)carve((size_t)nNodes * XSTR * 4);
  float* bufB  = (float*)carve((size_t)nNodes * XSTR * 4);
  float* m_i   = (float*)carve((size_t)nNodes * MD * 4);
  float* mhat  = (float*)carve((size_t)nNodes * 3 * 4);
  bf16*  w1t   = (bf16*)carve((size_t)(NT1 * 16) * KP1 * 2);
  bf16*  w2t   = (bf16*)carve((size_t)MD * KP2 * 2);
  bf16*  cw1t  = (bf16*)carve((size_t)256 * MD * 2);
  bf16*  nw1t  = (bf16*)carve((size_t)256 * KN1 * 2);
  bf16*  nw2t  = (bf16*)carve((size_t)FE * KN2 * 2);
  (void)ws_size;

  const int edgeGroups = (nEdges + 31) / 32;        // 10000 (32 edges / wave)
  const int edgeBlocks = (edgeGroups + 1) / 2;      // 5000  (2 waves / block)
  const int nodeTiles  = (nNodes + 15) / 16;        // 1250
  const int nodeBlocks = (nodeTiles + 1) / 2;       // 625

  auto prep = [&](const float* src, bf16* dst, int K, int N, int Kp, int Np) {
    int total = Np * Kp;
    egnn_prep_wT<<<(total + 255) / 256, 256, 0, stream>>>(src, dst, K, N, Kp, Np);
  };

  const float* xin = x;
  for (int l = 0; l < L; ++l) {
    float* xout = (l == 0) ? bufA : (l == 1) ? bufB : (float*)d_out;

    // weight prep (fp32 -> transposed/padded bf16) for this layer
    prep(ew1 + (size_t)l * EINK * N1, w1t, EINK, N1, KP1, NT1 * 16);
    prep(ew2 + (size_t)l * N1 * MD,   w2t, N1,  MD,  KP2, MD);
    prep(cw1 + (size_t)l * MD * 256,  cw1t, MD,  256, MD,  256);
    prep(nw1 + (size_t)l * KN1 * 256, nw1t, KN1, 256, KN1, 256);
    prep(nw2 + (size_t)l * KN2 * FE,  nw2t, KN2, FE,  KN2, FE);

    // zero accumulators for the segment sums
    {
      long n1 = (long)nNodes * MD;
      egnn_zero_f32<<<(int)((n1 + 255) / 256), 256, 0, stream>>>(m_i, n1);
      long n2 = (long)nNodes * 3;
      egnn_zero_f32<<<(int)((n2 + 255) / 256), 256, 0, stream>>>(mhat, n2);
    }

    egnn_edge_kernel<<<edgeBlocks, 64, 0, stream>>>(
        xin, ei, nEdges,
        w1t, eb1 + (size_t)l * N1,
        w2t, eb2 + (size_t)l * MD,
        cw1t, cb1 + (size_t)l * 256,
        cw2 + (size_t)l * 256, cb2 + (size_t)l,
        m_i, mhat);

    egnn_node_kernel<<<nodeBlocks, 64, 0, stream>>>(
        xin, m_i, mhat,
        nw1t, nb1 + (size_t)l * 256,
        nw2t, nb2 + (size_t)l * FE,
        ng + (size_t)l * FE, nb + (size_t)l * FE,
        xout, nNodes);

    xin = xout;
  }
}
